// MBV2_MCU_84146999263802
// MI455X (gfx1250) — compile-verified, matrix operand order checked by reading, not run on hardware
//
#include <hip/hip_runtime.h>
#include <hip/hip_bf16.h>
#include <math.h>

// ---------------------------------------------------------------------------
// MobileNetV2 prefix, batch 32 @ 224x224, fused BN + ReLU6, fp16 activations.
// Pointwise convs: v_wmma_f32_16x16x32_f16, A tiles staged to LDS via
// global_load_async_to_lds_b128 (+ s_wait_asynccnt).  CDNA5 gfx1250 wave32.
// ---------------------------------------------------------------------------

typedef __attribute__((ext_vector_type(16))) _Float16     v16h;
typedef __attribute__((ext_vector_type(8)))  float        v8f;
typedef __attribute__((ext_vector_type(4)))  unsigned int v4u;

union Frag16 { v16h h; uint32_t u[8]; };

#define BN_EPS 1e-3f

// ------------------------------ BN folding --------------------------------
// y = (conv - mean) * scale * rsqrt(var+eps) + offset  =  conv * g + b.
// Fold g into the weights (output channel = innermost dim of HWIO).

__global__ void fold_w_f16(const float* __restrict__ w,
                           const float* __restrict__ mean, const float* __restrict__ var,
                           const float* __restrict__ scale, const float* __restrict__ offset,
                           _Float16* __restrict__ wf, float* __restrict__ bias,
                           int total, int C) {
    int idx = blockIdx.x * blockDim.x + threadIdx.x;
    if (idx >= total) return;
    int c = idx % C;
    float g = scale[c] * rsqrtf(var[c] + BN_EPS);
    wf[idx] = (_Float16)(w[idx] * g);
    if (idx < C) bias[idx] = offset[idx] - mean[idx] * g;
}

__global__ void fold_w_f32(const float* __restrict__ w,
                           const float* __restrict__ mean, const float* __restrict__ var,
                           const float* __restrict__ scale, const float* __restrict__ offset,
                           float* __restrict__ wf, float* __restrict__ bias,
                           int total, int C) {
    int idx = blockIdx.x * blockDim.x + threadIdx.x;
    if (idx >= total) return;
    int c = idx % C;
    float g = scale[c] * rsqrtf(var[c] + BN_EPS);
    wf[idx] = w[idx] * g;
    if (idx < C) bias[idx] = offset[idx] - mean[idx] * g;
}

// ------------------------------ stem conv ---------------------------------
// 3x3 s2 SAME (pad_before=0), Cin=3, Cout=32. One thread per output pixel.

__global__ void __launch_bounds__(256)
stem_kernel(const float* __restrict__ x, const float* __restrict__ wf,
            const float* __restrict__ bias, _Float16* __restrict__ out) {
    int idx = blockIdx.x * blockDim.x + threadIdx.x;   // over 32*112*112
    const int total = 32 * 112 * 112;
    if (idx >= total) return;
    int ox = idx % 112;
    int oy = (idx / 112) % 112;
    int n  = idx / (112 * 112);

    float acc[32];
#pragma unroll
    for (int co = 0; co < 32; ++co) acc[co] = bias[co];

    for (int dy = 0; dy < 3; ++dy) {
        int iy = oy * 2 + dy;                 // pad_before = 0
        if (iy >= 224) continue;
        for (int dx = 0; dx < 3; ++dx) {
            int ix = ox * 2 + dx;
            if (ix >= 224) continue;
            const float* px = x + (((size_t)n * 224 + iy) * 224 + ix) * 3;
#pragma unroll
            for (int ci = 0; ci < 3; ++ci) {
                float v = px[ci];
                const float* wr = wf + ((dy * 3 + dx) * 3 + ci) * 32;
#pragma unroll
                for (int co = 0; co < 32; ++co) acc[co] = fmaf(v, wr[co], acc[co]);
            }
        }
    }
    _Float16* o = out + (size_t)idx * 32;
#pragma unroll
    for (int co = 0; co < 32; ++co)
        o[co] = (_Float16)fminf(fmaxf(acc[co], 0.f), 6.f);
}

// --------------------------- depthwise 3x3 --------------------------------

__global__ void __launch_bounds__(256)
dw_kernel(const _Float16* __restrict__ in, const float* __restrict__ wf,
          const float* __restrict__ bias, _Float16* __restrict__ out,
          int C, int Hin, int Win, int Hout, int Wout, int stride, int padBefore,
          int total) {
    int idx = blockIdx.x * blockDim.x + threadIdx.x;
    if (idx >= total) return;
    int c  = idx % C;
    int t  = idx / C;
    int ox = t % Wout; t /= Wout;
    int oy = t % Hout;
    int n  = t / Hout;

    float a = bias[c];
#pragma unroll
    for (int dy = 0; dy < 3; ++dy) {
        int iy = oy * stride + dy - padBefore;
        if ((unsigned)iy >= (unsigned)Hin) continue;
#pragma unroll
        for (int dx = 0; dx < 3; ++dx) {
            int ix = ox * stride + dx - padBefore;
            if ((unsigned)ix >= (unsigned)Win) continue;
            float v = (float)in[(((size_t)n * Hin + iy) * Win + ix) * C + c];
            a = fmaf(v, wf[(dy * 3 + dx) * C + c], a);
        }
    }
    out[idx] = (_Float16)fminf(fmaxf(a, 0.f), 6.f);
}

// ----------------------- WMMA pointwise-conv GEMM -------------------------
// out[M, COUT] = act(A[M, K] * W[K, COUT] + bias) (+ res).  256 thr = 8 waves.
// Wave -> one 16-row M tile, ceil(COUT/16) N tiles in register accumulators.
// A tile (16x32 f16 = 1KB) staged with global_load_async_to_lds_b128; K pad
// chunks zeroed in LDS (garbage fp16 could be NaN/Inf and WMMA propagates).
// W staged co-major, zero-padded to [CPAD][KPAD].  Requires M%128==0, K%8==0.

template <int K, int COUT, int RELU, int HASRES>
__global__ void __launch_bounds__(256)
gemm_wmma(const _Float16* __restrict__ Ain, const _Float16* __restrict__ W,
          const float* __restrict__ bias, const _Float16* __restrict__ res,
          _Float16* __restrict__ out) {
    constexpr int KPAD = ((K + 31) / 32) * 32;
    constexpr int CPAD = ((COUT + 15) / 16) * 16;
    constexpr int NT   = CPAD / 16;

    __shared__ alignas(16) _Float16 sW[CPAD * KPAD];   // sW[co*KPAD + k]
    __shared__ alignas(16) _Float16 sA[8 * 512];       // per-wave 16x32 tile

    const int tid  = threadIdx.x;
    const int lane = tid & 31;
    const int wave = tid >> 5;
    const int half = lane >> 4;     // 0: K {0..7,16..23}  1: K {8..15,24..31}
    const int nlo  = lane & 15;     // A row / B col / C col

    // Stage W once (co-major, zero pad in both K and COUT).
    for (int idx = tid; idx < CPAD * KPAD; idx += 256) {
        int co = idx / KPAD, k = idx % KPAD;
        sW[idx] = (co < COUT && k < K) ? W[(size_t)k * COUT + co] : (_Float16)0.f;
    }
    __syncthreads();

    v8f acc[NT];
#pragma unroll
    for (int nt = 0; nt < NT; ++nt)
#pragma unroll
        for (int i = 0; i < 8; ++i) acc[nt][i] = 0.f;

    const size_t rowbase = (size_t)blockIdx.x * 128 + (size_t)wave * 16;
    const uint32_t* myA  = (const uint32_t*)(sA + wave * 512);
    const uint32_t* sW32 = (const uint32_t*)sW;
    // Low 32 bits of a flat shared address are the LDS byte offset.
    const uint32_t sAoff = (uint32_t)(uintptr_t)(const void*)sA + (uint32_t)wave * 1024u;

    for (int kt = 0; kt < KPAD; kt += 32) {
        // --- stage this wave's 16x32 A tile: 64 x 16B chunks over 32 lanes ---
#pragma unroll
        for (int i = 0; i < 2; ++i) {
            int c = i * 32 + lane;          // chunk id: row r, quarter q
            int r = c >> 2, q = c & 3;
            int k = kt + q * 8;             // K % 8 == 0 -> chunk all-in or all-pad
            if (k < K) {
                uint64_t ga = (uint64_t)(uintptr_t)(Ain + (rowbase + r) * (size_t)K + k);
                uint32_t lo = sAoff + (uint32_t)c * 16u;
                asm volatile("global_load_async_to_lds_b128 %0, %1, off"
                             :: "v"(lo), "v"(ga) : "memory");
            } else {
                uint32_t* zp = (uint32_t*)(sA + wave * 512) + c * 4;
                zp[0] = 0u; zp[1] = 0u; zp[2] = 0u; zp[3] = 0u;
            }
        }
        if (kt + 32 < KPAD)   // hint next K tile -> global_prefetch_b8
            __builtin_prefetch((const void*)(Ain + (rowbase + nlo) * (size_t)K + kt + 32), 0, 1);
        asm volatile("s_wait_asynccnt 0x0" ::: "memory");
        __syncthreads();

        // --- A fragment: row = nlo; 8 K-pairs = 2 contiguous 16B groups ---
        Frag16 af;
        {
            v4u lo = *(const v4u*)(myA + nlo * 16 + (half << 2));
            v4u hi = *(const v4u*)(myA + nlo * 16 + 8 + (half << 2));
#pragma unroll
            for (int j = 0; j < 4; ++j) { af.u[j] = lo[j]; af.u[4 + j] = hi[j]; }
        }

#pragma unroll
        for (int nt = 0; nt < NT; ++nt) {
            int co = nt * 16 + nlo;
            Frag16 bf;
            {
                const uint32_t* wb = sW32 + ((co * KPAD + kt) >> 1);
                v4u lo = *(const v4u*)(wb + (half << 2));
                v4u hi = *(const v4u*)(wb + 8 + (half << 2));
#pragma unroll
                for (int j = 0; j < 4; ++j) { bf.u[j] = lo[j]; bf.u[4 + j] = hi[j]; }
            }
            acc[nt] = __builtin_amdgcn_wmma_f32_16x16x32_f16(
                false, af.h, false, bf.h, (short)0, acc[nt], false, false);
        }
        __syncthreads();
    }

    // Epilogue: C/D layout -> VGPR r: lanes0-15 M=r, lanes16-31 M=r+8.
    const int mofs = half << 3;
#pragma unroll
    for (int nt = 0; nt < NT; ++nt) {
        int co = nt * 16 + nlo;
        if (co < COUT) {
            float bv = bias[co];
#pragma unroll
            for (int r = 0; r < 8; ++r) {
                size_t row = rowbase + r + mofs;
                float v = acc[nt][r] + bv;
                if (RELU) v = fminf(fmaxf(v, 0.f), 6.f);
                if (HASRES) v += (float)res[row * COUT + co];
                out[row * COUT + co] = (_Float16)v;
            }
        }
    }
}

// ------------------------------ GAP ---------------------------------------

__global__ void __launch_bounds__(128)
gap_kernel(const _Float16* __restrict__ in, float* __restrict__ gap) {
    __shared__ float red[128];
    int b = blockIdx.x;          // 32*24
    int n = b / 24, c = b % 24;
    int tid = threadIdx.x;
    float s = 0.f;
    for (int p = tid; p < 3136; p += 128)
        s += (float)in[((size_t)n * 3136 + p) * 24 + c];
    red[tid] = s;
    __syncthreads();
    for (int w = 64; w > 0; w >>= 1) {
        if (tid < w) red[tid] += red[tid + w];
        __syncthreads();
    }
    if (tid == 0) gap[b] = red[0] * (1.f / 3136.f);
}

// --------------------------- FC + softmax ---------------------------------

__global__ void __launch_bounds__(256)
fc_softmax_kernel(const float* __restrict__ gap, const float* __restrict__ fw,
                  const float* __restrict__ fb, float* __restrict__ out) {
    __shared__ float sg[24];
    __shared__ float sl[1000];
    __shared__ float red[256];
    int n = blockIdx.x, tid = threadIdx.x;
    if (tid < 24) sg[tid] = gap[n * 24 + tid];
    __syncthreads();

    float lmax = -1e30f;
    for (int j = tid; j < 1000; j += 256) {
        float a = fb[j];
#pragma unroll
        for (int k = 0; k < 24; ++k) a = fmaf(sg[k], fw[k * 1000 + j], a);
        sl[j] = a;
        lmax = fmaxf(lmax, a);
    }
    red[tid] = lmax; __syncthreads();
    for (int w = 128; w > 0; w >>= 1) {
        if (tid < w) red[tid] = fmaxf(red[tid], red[tid + w]);
        __syncthreads();
    }
    float mx = red[0];
    __syncthreads();

    float lsum = 0.f;
    for (int j = tid; j < 1000; j += 256) {
        float e = expf(sl[j] - mx);
        sl[j] = e;
        lsum += e;
    }
    red[tid] = lsum; __syncthreads();
    for (int w = 128; w > 0; w >>= 1) {
        if (tid < w) red[tid] += red[tid + w];
        __syncthreads();
    }
    float inv = 1.f / red[0];
    for (int j = tid; j < 1000; j += 256)
        out[n * 1000 + j] = sl[j] * inv;
}

// ----------------------------- host side ----------------------------------

extern "C" void kernel_launch(void* const* d_in, const int* in_sizes, int n_in,
                              void* d_out, int out_size, void* d_ws, size_t ws_size,
                              hipStream_t stream) {
    (void)in_sizes; (void)n_in; (void)out_size; (void)ws_size;

    // Input map (setup_inputs dict order, BN = mean,var,scale,offset):
    const float* x     = (const float*)d_in[0];
    const float* stem_w = (const float*)d_in[1];
#define BN4(i) (const float*)d_in[(i)], (const float*)d_in[(i)+1], (const float*)d_in[(i)+2], (const float*)d_in[(i)+3]
    const float* dw1_w = (const float*)d_in[6];
    const float* pw1_w = (const float*)d_in[11];
    const float* exp2_w = (const float*)d_in[16];
    const float* dw2_w = (const float*)d_in[21];
    const float* pw2_w = (const float*)d_in[26];
    const float* exp3_w = (const float*)d_in[31];
    const float* dw3_w = (const float*)d_in[36];
    const float* pw3_w = (const float*)d_in[41];
    const float* fc_w  = (const float*)d_in[46];
    const float* fc_b  = (const float*)d_in[47];

    // Workspace carve.
    char* base = (char*)d_ws;
    size_t off = 0;
    auto carve = [&](size_t bytes) { char* p = base + off; off += (bytes + 255) & ~(size_t)255; return p; };

    const size_t BIG = 38535168;          // 32*112*112*96 (largest activation)
    _Float16* bufA   = (_Float16*)carve(BIG * 2);
    _Float16* bufB   = (_Float16*)carve(BIG * 2);
    _Float16* bufRes = (_Float16*)carve((size_t)2408448 * 2);   // 32*56*56*24

    _Float16* wpw1  = (_Float16*)carve(512  * 2);
    _Float16* wexp2 = (_Float16*)carve(1536 * 2);
    _Float16* wpw2  = (_Float16*)carve(2304 * 2);
    _Float16* wexp3 = (_Float16*)carve(3456 * 2);
    _Float16* wpw3  = (_Float16*)carve(3456 * 2);
    float* bpw1  = (float*)carve(16  * 4);
    float* bexp2 = (float*)carve(96  * 4);
    float* bpw2  = (float*)carve(24  * 4);
    float* bexp3 = (float*)carve(144 * 4);
    float* bpw3  = (float*)carve(24  * 4);
    float* wstem = (float*)carve(864 * 4);   float* bstem = (float*)carve(32 * 4);
    float* wdw1  = (float*)carve(288 * 4);   float* bdw1  = (float*)carve(32 * 4);
    float* wdw2  = (float*)carve(864 * 4);   float* bdw2  = (float*)carve(96 * 4);
    float* wdw3  = (float*)carve(1296 * 4);  float* bdw3  = (float*)carve(144 * 4);
    float* gapb  = (float*)carve(768 * 4);

    // --- BN folding (tiny) ---
    fold_w_f32<<<4, 256, 0, stream>>>(stem_w, BN4(2),  wstem, bstem, 864, 32);
    fold_w_f32<<<2, 256, 0, stream>>>(dw1_w,  BN4(7),  wdw1,  bdw1,  288, 32);
    fold_w_f16<<<2, 256, 0, stream>>>(pw1_w,  BN4(12), wpw1,  bpw1,  512, 16);
    fold_w_f16<<<6, 256, 0, stream>>>(exp2_w, BN4(17), wexp2, bexp2, 1536, 96);
    fold_w_f32<<<4, 256, 0, stream>>>(dw2_w,  BN4(22), wdw2,  bdw2,  864, 96);
    fold_w_f16<<<9, 256, 0, stream>>>(pw2_w,  BN4(27), wpw2,  bpw2,  2304, 24);
    fold_w_f16<<<14, 256, 0, stream>>>(exp3_w, BN4(32), wexp3, bexp3, 3456, 144);
    fold_w_f32<<<6, 256, 0, stream>>>(dw3_w,  BN4(37), wdw3,  bdw3,  1296, 144);

    // --- network ---
    stem_kernel<<<(401408 + 255) / 256, 256, 0, stream>>>(x, wstem, bstem, bufA);
    {
        int total = 32 * 112 * 112 * 32;      // dw1 s1 C=32: A -> B
        dw_kernel<<<(total + 255) / 256, 256, 0, stream>>>(bufA, wdw1, bdw1, bufB,
                                                           32, 112, 112, 112, 112, 1, 1, total);
    }
    // pw1: [401408,32]x[32,16] -> A
    gemm_wmma<32, 16, 0, 0><<<3136, 256, 0, stream>>>(bufB, wpw1, bpw1, nullptr, bufA);
    // exp2: [401408,16]x[16,96] -> B (relu6)
    gemm_wmma<16, 96, 1, 0><<<3136, 256, 0, stream>>>(bufA, wexp2, bexp2, nullptr, bufB);
    {
        int total = 32 * 56 * 56 * 96;        // dw2 s2 C=96: B(112) -> A(56)
        dw_kernel<<<(total + 255) / 256, 256, 0, stream>>>(bufB, wdw2, bdw2, bufA,
                                                           96, 112, 112, 56, 56, 2, 0, total);
    }
    // pw2: [100352,96]x[96,24] -> RES
    gemm_wmma<96, 24, 0, 0><<<784, 256, 0, stream>>>(bufA, wpw2, bpw2, nullptr, bufRes);
    // exp3: RES x [24,144] -> A (relu6)
    gemm_wmma<24, 144, 1, 0><<<784, 256, 0, stream>>>(bufRes, wexp3, bexp3, nullptr, bufA);
    {
        int total = 32 * 56 * 56 * 144;       // dw3 s1 C=144: A -> B
        dw_kernel<<<(total + 255) / 256, 256, 0, stream>>>(bufA, wdw3, bdw3, bufB,
                                                           144, 56, 56, 56, 56, 1, 1, total);
    }
    // pw3 + residual: [100352,144]x[144,24] + RES -> A
    gemm_wmma<144, 24, 0, 1><<<784, 256, 0, stream>>>(bufB, wpw3, bpw3, bufRes, bufA);

    gap_kernel<<<768, 128, 0, stream>>>(bufA, gapb);
    fc_softmax_kernel<<<32, 256, 0, stream>>>(gapb, fc_w, fc_b, (float*)d_out);
#undef BN4
}
